// LongformerAttentionMethod_60473139528446
// MI455X (gfx1250) — matrix operand order repaired and checked by reading, not compiled, not run on hardware
//
#include <hip/hip_runtime.h>

// Longformer sliding-window attention for MI455X (gfx1250).
// bf16 WMMA (16x16x32) for QK^T, P*V and row-sums; fp32 online softmax.

constexpr int L_SEQ  = 4096;
constexpr int D_HEAD = 64;
constexpr int WIN    = 256;
constexpr int H_NUM  = 12;
constexpr int BH_NUM = 24;                       // B*H
constexpr int QT_WG  = 128;                      // query rows per WG
constexpr int NWAVE  = QT_WG / 16;               // 8 waves
constexpr int NTHR   = NWAVE * 32;               // 256 threads
constexpr int K_TILE = 32;                       // keys per streamed tile
constexpr int SPAN   = QT_WG + 2 * WIN;          // 640-key union span
constexpr int N_KT   = SPAN / K_TILE;            // 20 tiles
constexpr float SCALE_LOG2 = 0.125f * 1.44269504088896340736f; // 1/sqrt(D) * log2(e)

typedef __attribute__((ext_vector_type(16))) __bf16         v16bf;
typedef __attribute__((ext_vector_type(8)))  float          v8f;
typedef __attribute__((ext_vector_type(8)))  unsigned short u16x8;

union Frag {
  v16bf v;
  u16x8 h[2];
};

struct alignas(8) bf16x4 { __bf16 a, b, c, d; };

// butterfly reduce step inside 16-lane halves: ds_swizzle group-of-32, xor_mask=M
template <int M>
__device__ __forceinline__ float swzx(float x) {
  constexpr int imm = (M << 10) | 0x1F;  // {xor=M, or=0, and=0x1f}
  return __builtin_bit_cast(float,
         __builtin_amdgcn_ds_swizzle(__builtin_bit_cast(int, x), imm));
}

// raw v_exp_f32: args are always <= 0 here, flush-to-zero underflow is desired
__device__ __forceinline__ float fast_exp2(float x) {
  return __builtin_amdgcn_exp2f(x);
}

__device__ __forceinline__ v8f wmma_bf16(const Frag& a, const Frag& b, v8f c) {
  return __builtin_amdgcn_wmma_f32_16x16x32_bf16(false, a.v, false, b.v,
                                                 (short)0, c, false, false);
}

__global__ __launch_bounds__(NTHR)
void longformer_wmma_kernel(const float* __restrict__ q,
                            const float* __restrict__ k,
                            const float* __restrict__ v,
                            const unsigned char* __restrict__ pm,
                            float* __restrict__ out) {
  __shared__ __align__(16) __bf16 Qs[QT_WG * D_HEAD];    // [row][d]
  __shared__ __align__(16) __bf16 Ks[K_TILE * D_HEAD];   // [key][d]
  __shared__ __align__(16) __bf16 Vs[D_HEAD * K_TILE];   // [d][key] (transposed)
  __shared__ __align__(16) __bf16 Pb[NWAVE * 16 * 32];   // per-wave probs
  __shared__ unsigned char Pms[SPAN];                    // pad/OOB validity per key

  const int tid  = threadIdx.x;
  const int lane = tid & 31;
  const int wv   = tid >> 5;
  const int hl   = lane >> 4;
  const int lr   = lane & 15;

  const int q0   = blockIdx.x * QT_WG;
  const int bh   = blockIdx.y;
  const int bidx = bh / H_NUM;
  const size_t base_qk = (size_t)bh * L_SEQ * D_HEAD;

  // ---- stage Q once: fp32 -> bf16, pre-scaled by (1/sqrt(D))*log2(e) ----
  for (int u = 0; u < 8; ++u) {
    int e   = (u * NTHR + tid) * 4;            // element in 128x64 tile
    int row = e >> 6, dc = e & 63;
    const float4 qv = *(const float4*)(q + base_qk + (size_t)(q0 + row) * D_HEAD + dc);
    bf16x4 pkd = { (__bf16)(qv.x * SCALE_LOG2), (__bf16)(qv.y * SCALE_LOG2),
                   (__bf16)(qv.z * SCALE_LOG2), (__bf16)(qv.w * SCALE_LOG2) };
    *(bf16x4*)(Qs + row * D_HEAD + dc) = pkd;
  }
  // ---- stage validity mask for the whole 640-key span ----
  for (int u = tid; u < SPAN; u += NTHR) {
    int j = q0 - WIN + u;
    Pms[u] = (j >= 0 && j < L_SEQ) ? pm[(size_t)bidx * L_SEQ + j] : (unsigned char)0;
  }
  __syncthreads();

  // ---- per-wave Q A-fragments: 16x64 as two 16x32 K-chunks ----
  Frag aq[2];
  {
    const unsigned short* qrow = (const unsigned short*)Qs + (wv * 16 + lr) * D_HEAD;
#pragma unroll
    for (int c = 0; c < 2; ++c) {
      aq[c].h[0] = *(const u16x8*)(qrow + c * 32 + hl * 16);
      aq[c].h[1] = *(const u16x8*)(qrow + c * 32 + hl * 16 + 8);
    }
  }
  Frag fOnes;
  {
    u16x8 o8 = (unsigned short)0x3F80;   // bf16(1.0) splat
    fOnes.h[0] = o8; fOnes.h[1] = o8;
  }

  const int q0w = q0 + wv * 16;
  float mrow[8], lrow[8];
  v8f acc[4];
#pragma unroll
  for (int r = 0; r < 8; ++r) { mrow[r] = -1e30f; lrow[r] = 0.0f; }
#pragma unroll
  for (int dt = 0; dt < 4; ++dt) acc[dt] = 0.0f;

  for (int kt = 0; kt < N_KT; ++kt) {
    const int jb = q0 - WIN + kt * K_TILE;
    __syncthreads();   // previous tile fully consumed

    // ---- stage K (row-major) / V (transposed) as bf16 ----
    for (int u = 0; u < 2; ++u) {
      int e   = (u * NTHR + tid) * 4;          // element in 32x64 tile
      int key = e >> 6, dc = e & 63;
      int j   = jb + key;
      float4 kv = make_float4(0.f, 0.f, 0.f, 0.f);
      float4 vv = make_float4(0.f, 0.f, 0.f, 0.f);
      if (j >= 0 && j < L_SEQ) {
        kv = *(const float4*)(k + base_qk + (size_t)j * D_HEAD + dc);
        vv = *(const float4*)(v + base_qk + (size_t)j * D_HEAD + dc);
      }
      int jn = j + K_TILE;
      if (kt + 1 < N_KT && jn >= 0 && jn < L_SEQ) {
        __builtin_prefetch(k + base_qk + (size_t)jn * D_HEAD + dc, 0, 1);
        __builtin_prefetch(v + base_qk + (size_t)jn * D_HEAD + dc, 0, 1);
      }
      bf16x4 kp = { (__bf16)kv.x, (__bf16)kv.y, (__bf16)kv.z, (__bf16)kv.w };
      *(bf16x4*)(Ks + key * D_HEAD + dc) = kp;
      Vs[(dc + 0) * K_TILE + key] = (__bf16)vv.x;
      Vs[(dc + 1) * K_TILE + key] = (__bf16)vv.y;
      Vs[(dc + 2) * K_TILE + key] = (__bf16)vv.z;
      Vs[(dc + 3) * K_TILE + key] = (__bf16)vv.w;
    }
    __syncthreads();

    // ---- scores: two 16x16 N-tiles; mask bias folded into C-init ----
    const int cbase = kt * K_TILE + lr;
    const float cb0 = Pms[cbase]      ? 0.0f : -1e30f;
    const float cb1 = Pms[cbase + 16] ? 0.0f : -1e30f;

    Frag kb00, kb01, kb10, kb11;
    {
      const unsigned short* kb = (const unsigned short*)Ks + lr * D_HEAD + hl * 16;
      kb00.h[0] = *(const u16x8*)(kb);        kb00.h[1] = *(const u16x8*)(kb + 8);
      kb01.h[0] = *(const u16x8*)(kb + 32);   kb01.h[1] = *(const u16x8*)(kb + 40);
      const unsigned short* kb2 = kb + 16 * D_HEAD;
      kb10.h[0] = *(const u16x8*)(kb2);       kb10.h[1] = *(const u16x8*)(kb2 + 8);
      kb11.h[0] = *(const u16x8*)(kb2 + 32);  kb11.h[1] = *(const u16x8*)(kb2 + 40);
    }
    v8f s0 = cb0;
    s0 = wmma_bf16(aq[0], kb00, s0);
    s0 = wmma_bf16(aq[1], kb01, s0);
    v8f s1 = cb1;
    s1 = wmma_bf16(aq[0], kb10, s1);
    s1 = wmma_bf16(aq[1], kb11, s1);

    // interior tile: window condition holds for every (row, col) of this wave
    const bool interior = (jb >= q0w + 15 - WIN) && (jb + K_TILE - 1 <= q0w + WIN) &&
                          (jb >= 0) && (jb + K_TILE <= L_SEQ);
    const int bwin = jb + lr - q0w - 8 * hl + WIN;   // per-lane window base (col nt0)
    __bf16* Pw = Pb + wv * 512;

    auto softmax_body = [&](bool MSK) {
#pragma unroll
      for (int r = 0; r < 8; ++r) {
        float sv0 = s0[r], sv1 = s1[r];
        if (MSK) {
          sv0 = ((unsigned)(bwin - r)      <= 2u * WIN) ? sv0 : -1e30f;
          sv1 = ((unsigned)(bwin + 16 - r) <= 2u * WIN) ? sv1 : -1e30f;
        }
        float rmax = fmaxf(sv0, sv1);                 // combined 32-col row max
        rmax = fmaxf(rmax, swzx<1>(rmax));
        rmax = fmaxf(rmax, swzx<2>(rmax));
        rmax = fmaxf(rmax, swzx<4>(rmax));
        rmax = fmaxf(rmax, swzx<8>(rmax));
        const float mnew = fmaxf(mrow[r], rmax);
        const float corr = fast_exp2(mrow[r] - mnew); // scores already in log2 domain
        mrow[r] = mnew;
        lrow[r] *= corr;
#pragma unroll
        for (int dt = 0; dt < 4; ++dt) acc[dt][r] *= corr;
        const float p0 = fast_exp2(sv0 - mnew);
        const float p1 = fast_exp2(sv1 - mnew);
        __bf16* pr = Pw + (r + 8 * hl) * 32 + lr;
        pr[0]  = (__bf16)p0;
        pr[16] = (__bf16)p1;
      }
    };
    if (interior) softmax_body(false); else softmax_body(true);

    // ---- P x V (4 WMMAs) + row-sum via P x ones (1 WMMA) ----
    Frag pa;
    {
      const unsigned short* prow = (const unsigned short*)Pw + lr * 32;
      pa.h[0] = *(const u16x8*)(prow + hl * 16);
      pa.h[1] = *(const u16x8*)(prow + hl * 16 + 8);
    }
    v8f rs = 0.0f;
    rs = wmma_bf16(pa, fOnes, rs);
#pragma unroll
    for (int dt = 0; dt < 4; ++dt) {
      Frag vb;
      const unsigned short* vbase = (const unsigned short*)Vs + (dt * 16 + lr) * K_TILE + hl * 16;
      vb.h[0] = *(const u16x8*)(vbase);
      vb.h[1] = *(const u16x8*)(vbase + 8);
      acc[dt] = wmma_bf16(pa, vb, acc[dt]);
    }
#pragma unroll
    for (int r = 0; r < 8; ++r) lrow[r] += rs[r];
  }

  // ---- epilogue: normalize, fp32 store ----
#pragma unroll
  for (int r = 0; r < 8; ++r) {
    const int i    = q0w + r + 8 * hl;
    const float nv = __builtin_amdgcn_rcpf(lrow[r]);  // lrow >= ~1 for valid rows
#pragma unroll
    for (int dt = 0; dt < 4; ++dt) {
      out[base_qk + (size_t)i * D_HEAD + dt * 16 + lr] = acc[dt][r] * nv;
    }
  }
}

extern "C" void kernel_launch(void* const* d_in, const int* in_sizes, int n_in,
                              void* d_out, int out_size, void* d_ws, size_t ws_size,
                              hipStream_t stream) {
  (void)in_sizes; (void)n_in; (void)out_size; (void)d_ws; (void)ws_size;
  const float* q  = (const float*)d_in[0];
  const float* k  = (const float*)d_in[1];
  const float* v  = (const float*)d_in[2];
  const unsigned char* pm = (const unsigned char*)d_in[3];
  float* out = (float*)d_out;

  dim3 grid(L_SEQ / QT_WG, BH_NUM);   // 32 x 24 workgroups
  dim3 block(NTHR);                   // 8 waves (wave32)
  hipLaunchKernelGGL(longformer_wmma_kernel, grid, block, 0, stream, q, k, v, pm, out);
}
